// KMeans_23098334118326
// MI455X (gfx1250) — matrix operand order repaired and checked by reading, not compiled, not run on hardware
//
#include <hip/hip_runtime.h>
#include <hip/hip_bf16.h>

typedef __attribute__((ext_vector_type(16))) __bf16 v16bf;
typedef __attribute__((ext_vector_type(8)))  float  v8f;
typedef __attribute__((ext_vector_type(4)))  int    v4i_;

#define DIM    128
#define KCENT  1024
#define NITERS 10

#if defined(__has_builtin)
#if __has_builtin(__builtin_amdgcn_global_load_async_to_lds_b128)
#define HAVE_ASYNC_LDS 1
#endif
#endif

typedef __attribute__((address_space(1))) v4i_ GV4;   // global int4 (clang: "int __vector(4) __device__ *")
typedef __attribute__((address_space(3))) v4i_ LV4;   // LDS int4

__device__ __forceinline__ GV4* to_global(const void* p) {
    return (GV4*)(unsigned long long)p;               // int->ptr cast crosses addrspace legally
}
__device__ __forceinline__ LV4* to_lds(void* p) {
    // generic LDS pointer: low 32 bits are the LDS byte offset
    return (LV4*)(unsigned)(unsigned long long)p;
}

__device__ __forceinline__ void async_wait0() {
#if defined(__has_builtin) && __has_builtin(__builtin_amdgcn_s_wait_asynccnt)
    __builtin_amdgcn_s_wait_asynccnt(0);
#else
    asm volatile("s_wait_asynccnt 0x0" ::: "memory");
#endif
}

// Split f32 into bf16 hi (exact truncation) + bf16 lo (rounded residual).
__device__ __forceinline__ void splitf(float x, __bf16& h, __bf16& l) {
    unsigned u  = __float_as_uint(x) & 0xffff0000u;
    float    hf = __uint_as_float(u);
    h = (__bf16)hf;          // exact: low mantissa already zero
    l = (__bf16)(x - hf);    // residual, |l| < 2^-7 * |x|
}

// ---- centroid L2 norms -----------------------------------------------------
__global__ void cnorm_kernel(const float* __restrict__ cent, float* __restrict__ cnorm) {
    int k = blockIdx.x * blockDim.x + threadIdx.x;
    if (k >= KCENT) return;
    const float* c = cent + (size_t)k * DIM;
    float s = 0.f;
    #pragma unroll 8
    for (int d = 0; d < DIM; ++d) { float v = c[d]; s = fmaf(v, v, s); }
    cnorm[k] = s;
}

// ---- pre-swizzle centroids into WMMA B-fragment order (hi/lo bf16) ---------
// B (32x16 bf16) layout: lane L -> column n=L%16; element e -> K = (L/16)*16 + e.
// Fragment array: byte offset = cb*4096 + kc*1024 + lane*32.
__global__ void make_cfrag_kernel(const float* __restrict__ cent,
                                  __bf16* __restrict__ c_hi,
                                  __bf16* __restrict__ c_lo) {
    int cb   = blockIdx.x;          // centroid block 0..63
    int kc   = threadIdx.x >> 5;    // K-chunk 0..3 (one wave each)
    int lane = threadIdx.x & 31;
    int n    = lane & 15;
    int hf   = lane >> 4;
    const float* crow = cent + (size_t)(cb * 16 + n) * DIM + kc * 32 + hf * 16;
    size_t fo = (((size_t)cb * 4 + kc) * 32 + lane) * 16;
    v16bf h, l;
    #pragma unroll
    for (int e = 0; e < 16; ++e) {
        __bf16 hv, lv; splitf(crow[e], hv, lv);
        h[e] = hv; l[e] = lv;
    }
    *(v16bf*)(c_hi + fo) = h;
    *(v16bf*)(c_lo + fo) = l;
}

// ---- assignment: WMMA distance GEMM + argmin -------------------------------
// Each wave owns one 16-row block of x; 8 waves per block share LDS-staged
// centroid B-fragments (double-buffered, 2 centroid blocks per phase, filled
// via async global->LDS loads tracked with ASYNCcnt).
__global__ void __launch_bounds__(256)
assign_kernel(const float* __restrict__ x,
              const __bf16* __restrict__ c_hi,
              const __bf16* __restrict__ c_lo,
              const float* __restrict__ cnorm,
              int* __restrict__ clusters, int nrb) {
    // per phase: 2 centroid blocks * 4 kc * 32 lanes * 16 elems = 4096 bf16 (8KB)
    __shared__ __align__(32) __bf16 lhi[2][4096];
    __shared__ __align__(32) __bf16 llo[2][4096];

    int t    = threadIdx.x;
    int wave = (int)blockIdx.x * 8 + (t >> 5);
    int lane = t & 31;
    int m    = lane & 15;
    int hf   = lane >> 4;

#if !defined(HAVE_ASYNC_LDS)
    v4i_ pend[4];
    char* pdh = nullptr;
    char* pdl = nullptr;
#endif
    // Stage centroid-block pair `cbp` into phase buffer `ph` (16KB / 256 thr = 64B each).
    auto stage_issue = [&](int cbp, int ph) {
        const char* gh = (const char*)c_hi + (size_t)cbp * 8192 + t * 32;
        const char* gl = (const char*)c_lo + (size_t)cbp * 8192 + t * 32;
        char* dh = (char*)&lhi[ph][0] + t * 32;
        char* dl = (char*)&llo[ph][0] + t * 32;
#if defined(HAVE_ASYNC_LDS)
        __builtin_amdgcn_global_load_async_to_lds_b128(to_global(gh),      to_lds(dh),      0, 0);
        __builtin_amdgcn_global_load_async_to_lds_b128(to_global(gh + 16), to_lds(dh + 16), 0, 0);
        __builtin_amdgcn_global_load_async_to_lds_b128(to_global(gl),      to_lds(dl),      0, 0);
        __builtin_amdgcn_global_load_async_to_lds_b128(to_global(gl + 16), to_lds(dl + 16), 0, 0);
#else
        pend[0] = *(const v4i_*)gh;
        pend[1] = *(const v4i_*)(gh + 16);
        pend[2] = *(const v4i_*)gl;
        pend[3] = *(const v4i_*)(gl + 16);
        pdh = dh; pdl = dl;
#endif
    };
    auto stage_wait = [&]() {
#if defined(HAVE_ASYNC_LDS)
        async_wait0();
#else
        if (pdh) {
            *(v4i_*)pdh        = pend[0];
            *(v4i_*)(pdh + 16) = pend[1];
            *(v4i_*)pdl        = pend[2];
            *(v4i_*)(pdl + 16) = pend[3];
            pdh = nullptr;
        }
#endif
    };

    // Build A fragments (hi/lo) once, reused across all 64 centroid blocks.
    // A (16x32 bf16): lane L -> row m=L%16; element e -> K=K0+(L/16)*16+e.
    const float* xrow = x + ((size_t)wave * 16 + m) * DIM;
    v16bf ah[4], al[4];
    #pragma unroll
    for (int kc = 0; kc < 4; ++kc) {
        float rv[16];
        const float* p0 = xrow + kc * 32 + hf * 16;
        *(float4*)&rv[0]  = *(const float4*)(p0);
        *(float4*)&rv[4]  = *(const float4*)(p0 + 4);
        *(float4*)&rv[8]  = *(const float4*)(p0 + 8);
        *(float4*)&rv[12] = *(const float4*)(p0 + 12);
        #pragma unroll
        for (int e = 0; e < 16; ++e) {
            __bf16 hv, lv; splitf(rv[e], hv, lv);
            ah[kc][e] = hv; al[kc][e] = lv;
        }
    }

    float mval[8]; int midx[8];
    #pragma unroll
    for (int r = 0; r < 8; ++r) { mval[r] = __builtin_inff(); midx[r] = 0x7fffffff; }
    int nlane = lane & 15;

    stage_issue(0, 0);
    stage_wait();
    __syncthreads();

    for (int cbp = 0; cbp < KCENT / 32; ++cbp) {     // 32 pairs of centroid blocks
        int ph = cbp & 1;
        if (cbp + 1 < KCENT / 32) stage_issue(cbp + 1, ph ^ 1);

        // Two independent accumulator chains (cb0 = 2*cbp, cb1 = 2*cbp+1).
        v8f acc0 = {}, acc1 = {};
        #pragma unroll
        for (int kc = 0; kc < 4; ++kc) {
            v16bf b0h = *(const v16bf*)&lhi[ph][kc * 512 + lane * 16];
            v16bf b0l = *(const v16bf*)&llo[ph][kc * 512 + lane * 16];
            v16bf b1h = *(const v16bf*)&lhi[ph][2048 + kc * 512 + lane * 16];
            v16bf b1l = *(const v16bf*)&llo[ph][2048 + kc * 512 + lane * 16];
            // f32-emulated dot: xh*ch + xl*ch + xh*cl (xl*cl term negligible)
            acc0 = __builtin_amdgcn_wmma_f32_16x16x32_bf16(false, ah[kc], false, b0h,
                                                           (short)0, acc0, false, false);
            acc1 = __builtin_amdgcn_wmma_f32_16x16x32_bf16(false, ah[kc], false, b1h,
                                                           (short)0, acc1, false, false);
            acc0 = __builtin_amdgcn_wmma_f32_16x16x32_bf16(false, al[kc], false, b0h,
                                                           (short)0, acc0, false, false);
            acc1 = __builtin_amdgcn_wmma_f32_16x16x32_bf16(false, al[kc], false, b1h,
                                                           (short)0, acc1, false, false);
            acc0 = __builtin_amdgcn_wmma_f32_16x16x32_bf16(false, ah[kc], false, b0l,
                                                           (short)0, acc0, false, false);
            acc1 = __builtin_amdgcn_wmma_f32_16x16x32_bf16(false, ah[kc], false, b1l,
                                                           (short)0, acc1, false, false);
        }

        int   cb0 = 2 * cbp, cb1 = 2 * cbp + 1;
        float cn0 = cnorm[cb0 * 16 + nlane];
        float cn1 = cnorm[cb1 * 16 + nlane];
        int   id0 = cb0 * 16 + nlane;
        int   id1 = cb1 * 16 + nlane;
        #pragma unroll
        for (int r = 0; r < 8; ++r) {
            float s0 = fmaf(-2.f, acc0[r], cn0);   // ||c||^2 - 2 x.c (x^2 const per row)
            if (s0 < mval[r]) { mval[r] = s0; midx[r] = id0; }
            float s1 = fmaf(-2.f, acc1[r], cn1);
            if (s1 < mval[r]) { mval[r] = s1; midx[r] = id1; }
        }

        stage_wait();
        __syncthreads();
    }

    // Butterfly argmin across the 16 lanes of each half (C layout: half h holds M = h*8 + r).
    #pragma unroll
    for (int off = 8; off; off >>= 1) {
        #pragma unroll
        for (int r = 0; r < 8; ++r) {
            float ov = __shfl_xor(mval[r], off, 32);
            int   oi = __shfl_xor(midx[r], off, 32);
            if (ov < mval[r] || (ov == mval[r] && oi < midx[r])) { mval[r] = ov; midx[r] = oi; }
        }
    }
    if ((lane == 0 || lane == 16) && wave < nrb) {
        int base = wave * 16 + hf * 8;
        #pragma unroll
        for (int r = 0; r < 8; ++r) clusters[base + r] = midx[r];
    }
}

// ---- deterministic segment-sum + centroid update (one block per centroid) --
__global__ void update_kernel(const int* __restrict__ clusters,
                              const float* __restrict__ x,
                              float* __restrict__ cent,
                              float* __restrict__ counts, int N) {
    int k = blockIdx.x;
    int d = threadIdx.x;           // one dim per thread, DIM threads
    float sum = 0.f; int cnt = 0;
    const v4i_* cl4 = (const v4i_*)clusters;
    int n4 = N >> 2;
    for (int i = 0; i < n4; ++i) {
        v4i_ c = cl4[i];
        size_t b = (size_t)i * 4 * DIM + d;
        if (c.x == k) { sum += x[b];           ++cnt; }
        if (c.y == k) { sum += x[b + DIM];     ++cnt; }
        if (c.z == k) { sum += x[b + 2 * DIM]; ++cnt; }
        if (c.w == k) { sum += x[b + 3 * DIM]; ++cnt; }
    }
    cent[(size_t)k * DIM + d] = sum / (float)cnt;   // 0/0 -> NaN, matches reference
    if (d == 0) counts[k] = (float)cnt;
}

// ---- pack outputs: [clusters | centroids | counts] as float ----------------
__global__ void emit_kernel(const int* __restrict__ clusters,
                            const float* __restrict__ cent,
                            const float* __restrict__ counts,
                            float* __restrict__ out, int N, int out_size) {
    int i = blockIdx.x * blockDim.x + threadIdx.x;
    if (i >= out_size) return;
    int o1 = N, o2 = N + KCENT * DIM, o3 = N + KCENT * DIM + KCENT;
    if (i < o1)      out[i] = (float)clusters[i];
    else if (i < o2) out[i] = cent[i - o1];
    else if (i < o3) out[i] = counts[i - o2];
}

extern "C" void kernel_launch(void* const* d_in, const int* in_sizes, int n_in,
                              void* d_out, int out_size, void* d_ws, size_t ws_size,
                              hipStream_t stream) {
    const float* x = (const float*)d_in[0];
    int N = in_sizes[0] / DIM;     // 131072 (k=1024, niters=10 fixed by problem instance)

    char* ws = (char*)d_ws;
    size_t off = 0;
    auto alloc = [&](size_t bytes) {
        char* p = ws + off;
        off = (off + bytes + 255) & ~(size_t)255;
        return p;
    };
    int*    clusters = (int*)   alloc((size_t)N * 4);
    float*  cent     = (float*) alloc((size_t)KCENT * DIM * 4);
    float*  counts   = (float*) alloc((size_t)KCENT * 4);
    float*  cnorm    = (float*) alloc((size_t)KCENT * 4);
    __bf16* c_hi     = (__bf16*)alloc((size_t)KCENT * DIM * 2);
    __bf16* c_lo     = (__bf16*)alloc((size_t)KCENT * DIM * 2);
    (void)ws_size; (void)n_in;

    // c0 = x[:K]
    (void)hipMemcpyAsync(cent, x, (size_t)KCENT * DIM * 4, hipMemcpyDeviceToDevice, stream);

    int nrb = N / 16;              // 8192 row blocks, one per wave
    for (int it = 0; it < NITERS; ++it) {
        cnorm_kernel     <<<(KCENT + 255) / 256, 256, 0, stream>>>(cent, cnorm);
        make_cfrag_kernel<<<KCENT / 16, 128, 0, stream>>>(cent, c_hi, c_lo);
        assign_kernel    <<<nrb / 8, 256, 0, stream>>>(x, c_hi, c_lo, cnorm, clusters, nrb);
        update_kernel    <<<KCENT, DIM, 0, stream>>>(clusters, x, cent, counts, N);
    }
    int total = N + KCENT * DIM + KCENT;
    int nout  = total < out_size ? total : out_size;
    emit_kernel<<<(nout + 255) / 256, 256, 0, stream>>>(clusters, cent, counts,
                                                        (float*)d_out, N, nout);
}
